// VolumeNormalizer_2156073582850
// MI455X (gfx1250) — compile-verified, matrix operand order checked by reading, not run on hardware
//
#include <hip/hip_runtime.h>
#include <hip/hip_bf16.h>

typedef __attribute__((ext_vector_type(16))) _Float16 v16h;
typedef __attribute__((ext_vector_type(8)))  _Float16 v8h;
typedef __attribute__((ext_vector_type(8)))  float    v8f;

#define BB   32
#define NN   12000
#define TT   20000
#define NN1  4000
#define NN2  1000
#define N12  5000
#define N12P 5008
#define SPLITS 8
#define TBLK 79      // ceil(TT/256)

// ---------------------------------------------------------------------------
// Build pts[b][i][0..2]: x overrides rows [0,4000), y overrides x/z of rows
// [4000,5000), else points_zero.
__global__ void k_pts(const float* __restrict__ x, const float* __restrict__ y,
                      const float* __restrict__ pz, float* __restrict__ pts) {
  int idx = blockIdx.x * 256 + threadIdx.x;
  if (idx >= BB * NN) return;
  int b = idx / NN, i = idx % NN;
  float p0, p1, p2;
  if (i < NN1) {
    const float* xr = x + ((size_t)b * NN1 + i) * 3;
    p0 = xr[0]; p1 = xr[1]; p2 = xr[2];
  } else if (i < N12) {
    int j = i - NN1;
    const float* yr = y + ((size_t)b * NN2 + j) * 2;
    p0 = yr[0]; p1 = pz[i * 3 + 1]; p2 = yr[1];
  } else {
    p0 = pz[i * 3]; p1 = pz[i * 3 + 1]; p2 = pz[i * 3 + 2];
  }
  float* o = pts + ((size_t)b * NN + i) * 3;
  o[0] = p0; o[1] = p1; o[2] = p2;
}

// ---------------------------------------------------------------------------
// vol_const partials from base points (one partial per block).
__global__ void k_volc(const float* __restrict__ pz, const int* __restrict__ tri,
                       float* __restrict__ volc_part) {
  __shared__ float sm[256];
  int t = blockIdx.x * 256 + threadIdx.x;
  float v = 0.f;
  if (t < TT) {
    int i0 = tri[t * 3 + 0], i1 = tri[t * 3 + 1], i2 = tri[t * 3 + 2];
    float x0 = pz[i0 * 3], y0 = pz[i0 * 3 + 1], z0 = pz[i0 * 3 + 2];
    float x1 = pz[i1 * 3], y1 = pz[i1 * 3 + 1], z1 = pz[i1 * 3 + 2];
    float x2 = pz[i2 * 3], y2 = pz[i2 * 3 + 1], z2 = pz[i2 * 3 + 2];
    float det = ((x0 - x1) * (z2 - z1) - (z0 - z1) * (x2 - x1)) * (1.0f / 6.0f);
    v = (y0 + y1 + y2) * det;
  }
  sm[threadIdx.x] = v; __syncthreads();
  for (int s = 128; s > 0; s >>= 1) {
    if (threadIdx.x < s) sm[threadIdx.x] += sm[threadIdx.x + s];
    __syncthreads();
  }
  if (threadIdx.x == 0) volc_part[blockIdx.x] = sm[0];
}

// ---------------------------------------------------------------------------
// dy16[b][t] = det_y (stored f16 for direct WMMA A-fragment loads);
// also per-block partial of sum(y)*dy (f32).
__global__ void k_dy(const float* __restrict__ pts, const int* __restrict__ tri,
                     _Float16* __restrict__ dy16, float* __restrict__ vol_part) {
  __shared__ float sm[256];
  int t = blockIdx.x * 256 + threadIdx.x;
  int b = blockIdx.y;
  const float* P = pts + (size_t)b * NN * 3;
  float contrib = 0.f;
  if (t < TT) {
    int i0 = tri[t * 3 + 0], i1 = tri[t * 3 + 1], i2 = tri[t * 3 + 2];
    float x0 = P[i0 * 3], y0 = P[i0 * 3 + 1], z0 = P[i0 * 3 + 2];
    float x1 = P[i1 * 3], y1 = P[i1 * 3 + 1], z1 = P[i1 * 3 + 2];
    float x2 = P[i2 * 3], y2 = P[i2 * 3 + 1], z2 = P[i2 * 3 + 2];
    float det = ((x0 - x1) * (z2 - z1) - (z0 - z1) * (x2 - x1)) * (1.0f / 6.0f);
    dy16[(size_t)b * TT + t] = (_Float16)det;
    contrib = (y0 + y1 + y2) * det;
  }
  sm[threadIdx.x] = contrib; __syncthreads();
  for (int s = 128; s > 0; s >>= 1) {
    if (threadIdx.x < s) sm[threadIdx.x] += sm[threadIdx.x + s];
    __syncthreads();
  }
  if (threadIdx.x == 0) vol_part[b * TBLK + blockIdx.x] = sm[0];
}

// ---------------------------------------------------------------------------
// dz16[b][t] = det_z (uses x,y coords). Run after y-correction.
__global__ void k_dz(const float* __restrict__ pts, const int* __restrict__ tri,
                     _Float16* __restrict__ dz16) {
  int t = blockIdx.x * 256 + threadIdx.x;
  int b = blockIdx.y;
  if (t >= TT) return;
  const float* P = pts + (size_t)b * NN * 3;
  int i0 = tri[t * 3 + 0], i1 = tri[t * 3 + 1], i2 = tri[t * 3 + 2];
  float x0 = P[i0 * 3], y0 = P[i0 * 3 + 1];
  float x1 = P[i1 * 3], y1 = P[i1 * 3 + 1];
  float x2 = P[i2 * 3], y2 = P[i2 * 3 + 1];
  float det = ((x0 - x2) * (y1 - y2) - (y0 - y2) * (x1 - x2)) * (1.0f / 6.0f);
  dz16[(size_t)b * TT + t] = (_Float16)det;
}

// ---------------------------------------------------------------------------
// Reduce vol_const & per-batch vol, produce a[b] = 0.5*(vol_const - vol[b]).
__global__ void k_red1(const float* __restrict__ volc_part,
                       const float* __restrict__ vol_part, float* __restrict__ a) {
  __shared__ float sm[128];
  float v = 0.f;
  for (int i = threadIdx.x; i < TBLK; i += 128) v += volc_part[i];
  sm[threadIdx.x] = v; __syncthreads();
  for (int s = 64; s > 0; s >>= 1) {
    if (threadIdx.x < s) sm[threadIdx.x] += sm[threadIdx.x + s];
    __syncthreads();
  }
  float volc = sm[0];
  __syncthreads();
  if (threadIdx.x < BB) {
    float s = 0.f;
    for (int i = 0; i < TBLK; i++) s += vol_part[threadIdx.x * TBLK + i];
    a[threadIdx.x] = 0.5f * (volc - s);
  }
}

// ---------------------------------------------------------------------------
// Fragment loaders. Per-lane K runs [ka, ka+8) and [ka+16, ka+24) per the
// gfx1250 16-bit operand layout (lane<16: K in [kg*8, kg*8+8) U [16+kg*8, ...)).
__device__ __forceinline__ v16h load_frag_h(const _Float16* __restrict__ row,
                                            int ka) {
  v8h lo = *(const v8h*)(row + ka);        // 1x b128
  v8h hi = *(const v8h*)(row + ka + 16);   // 1x b128
  v16h r;
#pragma unroll
  for (int i = 0; i < 8; i++) { r[i] = lo[i]; r[i + 8] = hi[i]; }
  return r;
}

__device__ __forceinline__ v16h load_frag_f(const float* __restrict__ row,
                                            int ka) {
  float4 f0 = *(const float4*)(row + ka);
  float4 f1 = *(const float4*)(row + ka + 4);
  float4 f2 = *(const float4*)(row + ka + 16);
  float4 f3 = *(const float4*)(row + ka + 20);
  v16h r;
  r[0]  = (_Float16)f0.x; r[1]  = (_Float16)f0.y;
  r[2]  = (_Float16)f0.z; r[3]  = (_Float16)f0.w;
  r[4]  = (_Float16)f1.x; r[5]  = (_Float16)f1.y;
  r[6]  = (_Float16)f1.z; r[7]  = (_Float16)f1.w;
  r[8]  = (_Float16)f2.x; r[9]  = (_Float16)f2.y;
  r[10] = (_Float16)f2.z; r[11] = (_Float16)f2.w;
  r[12] = (_Float16)f3.x; r[13] = (_Float16)f3.y;
  r[14] = (_Float16)f3.z; r[15] = (_Float16)f3.w;
  return r;
}

// C(32 x Ncols) = A16(32 x TT) * Bm(Ncols x TT)^T, split-K partials.
// One wave per (n-tile, k-split): two 16x16 accumulators share one B fragment,
// so each element of the 720MB streamed f32 matrices is read exactly once and
// converted to f16 in registers. Out-of-range columns of the last tile produce
// garbage only in their own output columns, which land in the Cpart padding
// and are never read — no masking needed (row pointer clamped for safety).
__global__ void __launch_bounds__(32)
k_gemm(const _Float16* __restrict__ A16, const float* __restrict__ Bm,
       float* __restrict__ Cpart, int Ncols, int NcolsPad) {
  const int lane  = threadIdx.x;
  const int mrow  = lane & 15;
  const int kg    = lane >> 4;
  const int ntile = blockIdx.x;
  const int s     = blockIdx.y;
  const int kChunk = ((TT / 32) + SPLITS - 1) / SPLITS;   // 79 wmma steps
  int k0 = s * kChunk * 32;
  int k1 = k0 + kChunk * 32; if (k1 > TT) k1 = TT;

  const int n = ntile * 16 + mrow;
  const _Float16* arow0 = A16 + (size_t)mrow * TT;          // batches 0..15
  const _Float16* arow1 = A16 + (size_t)(mrow + 16) * TT;   // batches 16..31
  const float* brow = Bm + (size_t)(n < Ncols ? n : (Ncols - 1)) * TT;

  v8f acc0 = {}; v8f acc1 = {};
  for (int k = k0; k < k1; k += 32) {
    int ka = k + kg * 8;
    __builtin_prefetch(brow + ka + 512, 0, 1);          // global_prefetch_b8
    v16h af0 = load_frag_h(arow0, ka);
    v16h af1 = load_frag_h(arow1, ka);
    v16h bf  = load_frag_f(brow, ka);
    acc0 = __builtin_amdgcn_wmma_f32_16x16x32_f16(false, af0, false, bf,
                                                  (short)0, acc0, false, false);
    acc1 = __builtin_amdgcn_wmma_f32_16x16x32_f16(false, af1, false, bf,
                                                  (short)0, acc1, false, false);
  }
  // D layout: lane -> N = lane&15, rows M = v + 8*(lane>>4)
  int cn = ntile * 16 + (lane & 15);
  float* base = Cpart + (size_t)s * BB * NcolsPad + cn;
#pragma unroll
  for (int v = 0; v < 8; v++) {
    int m = v + 8 * kg;
    base[(size_t)m * NcolsPad]        = acc0[v];
    base[(size_t)(m + 16) * NcolsPad] = acc1[v];
  }
}

// ---------------------------------------------------------------------------
// Deterministic split-K reduction.
__global__ void k_redC(const float* __restrict__ Cpart, float* __restrict__ C,
                       int Ncols, int NcolsPad) {
  int idx = blockIdx.x * 256 + threadIdx.x;
  if (idx >= BB * Ncols) return;
  int b = idx / Ncols, nn = idx % Ncols;
  float s = 0.f;
  for (int sp = 0; sp < SPLITS; sp++)
    s += Cpart[(size_t)sp * BB * NcolsPad + (size_t)b * NcolsPad + nn];
  C[(size_t)b * Ncols + nn] = s;
}

// Per-batch sum of squares of coeff rows.
__global__ void k_s2(const float* __restrict__ C, float* __restrict__ s2,
                     int Ncols) {
  __shared__ float sm[256];
  int b = blockIdx.x;
  float v = 0.f;
  for (int i = threadIdx.x; i < Ncols; i += 256) {
    float c = C[(size_t)b * Ncols + i];
    v += c * c;
  }
  sm[threadIdx.x] = v; __syncthreads();
  for (int s = 128; s > 0; s >>= 1) {
    if (threadIdx.x < s) sm[threadIdx.x] += sm[threadIdx.x + s];
    __syncthreads();
  }
  if (threadIdx.x == 0) s2[b] = sm[0];
}

// pts[b][i][coord] += coeff[b][i] * a[b] / s2[b]
__global__ void k_corr(float* __restrict__ pts, const float* __restrict__ C,
                       const float* __restrict__ a, const float* __restrict__ s2,
                       int count, int coord) {
  int idx = blockIdx.x * 256 + threadIdx.x;
  if (idx >= BB * count) return;
  int b = idx / count, i = idx % count;
  float scale = a[b] / s2[b];
  pts[((size_t)b * NN + i) * 3 + coord] += C[(size_t)b * count + i] * scale;
}

// out1 = pts[:, :4000, :], out2 = pts[:, 4000:5000, (0,2)], concatenated flat.
__global__ void k_out(const float* __restrict__ pts, float* __restrict__ out) {
  int idx = blockIdx.x * 256 + threadIdx.x;
  const int n1tot = BB * NN1 * 3;   // 384000
  const int n2tot = BB * NN2 * 2;   // 64000
  if (idx < n1tot) {
    int b = idx / (NN1 * 3);
    int r = idx % (NN1 * 3);
    int i = r / 3, c = r % 3;
    out[idx] = pts[((size_t)b * NN + i) * 3 + c];
  } else if (idx < n1tot + n2tot) {
    int j = idx - n1tot;
    int b = j / (NN2 * 2);
    int r = j % (NN2 * 2);
    int i = NN1 + r / 2;
    int c = (r % 2) * 2;   // coord 0 or 2
    out[idx] = pts[((size_t)b * NN + i) * 3 + c];
  }
}

// ---------------------------------------------------------------------------
extern "C" void kernel_launch(void* const* d_in, const int* in_sizes, int n_in,
                              void* d_out, int out_size, void* d_ws, size_t ws_size,
                              hipStream_t stream) {
  const float* x    = (const float*)d_in[0];   // (32, 12000)
  const float* y    = (const float*)d_in[1];   // (32, 2000)
  const float* pz   = (const float*)d_in[2];   // (12000, 3)
  const int*   tri  = (const int*)d_in[3];     // (20000, 3)
  const float* vfx  = (const float*)d_in[6];   // (4000, 20000)
  const float* vfxy = (const float*)d_in[7];   // (5000, 20000)
  float* out = (float*)d_out;

  float* w = (float*)d_ws;
  size_t o = 0;
  float* pts       = w + o; o += (size_t)BB * NN * 3;        // 1,152,000
  _Float16* dy16   = (_Float16*)(w + o); o += (size_t)BB * TT / 2;  // 320,000 f32 slots
  _Float16* dz16   = (_Float16*)(w + o); o += (size_t)BB * TT / 2;
  float* volc_part = w + o; o += 128;
  float* vol_part  = w + o; o += (size_t)BB * TBLK;          //     2,528
  float* a_        = w + o; o += BB;
  float* cyp       = w + o; o += (size_t)SPLITS * BB * NN1;  // 1,024,000
  float* czp       = w + o; o += (size_t)SPLITS * BB * N12P; // 1,282,048
  float* coeffy    = w + o; o += (size_t)BB * NN1;
  float* coeffz    = w + o; o += (size_t)BB * N12;
  float* s2y       = w + o; o += BB;
  float* s2z       = w + o; o += BB;

  // 1) pts build, vol_const partials
  k_pts<<<(BB * NN + 255) / 256, 256, 0, stream>>>(x, y, pz, pts);
  k_volc<<<TBLK, 256, 0, stream>>>(pz, tri, volc_part);

  // 2) dy (f16) + per-batch volume partials, then a[b]
  k_dy<<<dim3(TBLK, BB), 256, 0, stream>>>(pts, tri, dy16, vol_part);
  k_red1<<<1, 128, 0, stream>>>(volc_part, vol_part, a_);

  // 3) coeffy = dy @ vfx^T  (WMMA f16->f32, split-K)
  k_gemm<<<dim3(NN1 / 16, SPLITS), 32, 0, stream>>>(dy16, vfx, cyp, NN1, NN1);
  k_redC<<<(BB * NN1 + 255) / 256, 256, 0, stream>>>(cyp, coeffy, NN1, NN1);
  k_s2<<<BB, 256, 0, stream>>>(coeffy, s2y, NN1);
  k_corr<<<(BB * NN1 + 255) / 256, 256, 0, stream>>>(pts, coeffy, a_, s2y, NN1, 1);

  // 4) dz on corrected pts, coeffz = dz @ vfxy^T
  k_dz<<<dim3(TBLK, BB), 256, 0, stream>>>(pts, tri, dz16);
  k_gemm<<<dim3(N12P / 16, SPLITS), 32, 0, stream>>>(dz16, vfxy, czp, N12, N12P);
  k_redC<<<(BB * N12 + 255) / 256, 256, 0, stream>>>(czp, coeffz, N12, N12P);
  k_s2<<<BB, 256, 0, stream>>>(coeffz, s2z, N12);
  k_corr<<<(BB * N12 + 255) / 256, 256, 0, stream>>>(pts, coeffz, a_, s2z, N12, 2);

  // 5) gather outputs
  int ototal = BB * NN1 * 3 + BB * NN2 * 2;   // 448,000
  k_out<<<(ototal + 255) / 256, 256, 0, stream>>>(pts, out);
}
